// RVIT_Salinas_75505525063783
// MI455X (gfx1250) — matrix operand order repaired and checked by reading, not compiled, not run on hardware
//
#include <hip/hip_runtime.h>

// ---------------------------------------------------------------------------
// RVIT-Salinas forward pass for gfx1250 (MI455X).
// All GEMM-shaped ops lower to v_wmma_f32_16x16x32_f16 (f32 accumulate).
// Convs: unified implicit-GEMM kernel, 9 accumulated per-tap GEMMs whose
// B operand is staged zero-padded in LDS (unconditional ds_load fetches).
// Recurrent cells live entirely in LDS; one kernel per recurrent stage.
// ---------------------------------------------------------------------------

typedef __attribute__((ext_vector_type(16))) _Float16 v16h;
typedef __attribute__((ext_vector_type(8)))  float    v8f;

#define EPSF 1e-5f

// ---- wave-level 16x16 WMMA tile helpers (CDNA5 fragment layouts) ----------
template <class FA, class FB>
__device__ __forceinline__ v8f wmma_mac(v8f c, int K, FA fa, FB fb) {
  const int lane = threadIdx.x & 31;
  const int half = lane >> 4;
  const int l16  = lane & 15;
  for (int k0 = 0; k0 < K; k0 += 32) {
    v16h Af, Bf;
#pragma unroll
    for (int v = 0; v < 8; ++v) {
      const int ka = k0 + ((v & 4) ? 16 : 0) + half * 8 + (v & 3) * 2;
      Af[2 * v]     = (_Float16)fa(l16, ka);
      Af[2 * v + 1] = (_Float16)fa(l16, ka + 1);
      const int kb = k0 + ((v & 4) ? 16 : 0) + half * 8 + (v & 3) * 2;
      Bf[2 * v]     = (_Float16)fb(kb, l16);
      Bf[2 * v + 1] = (_Float16)fb(kb + 1, l16);
    }
    c = __builtin_amdgcn_wmma_f32_16x16x32_f16(false, Af, false, Bf,
                                               (short)0, c, false, false);
  }
  return c;
}

template <class FA, class FB>
__device__ __forceinline__ v8f wmma_acc(int K, FA fa, FB fb) {
  v8f c = {};
  return wmma_mac(c, K, fa, fb);
}

template <class FS>
__device__ __forceinline__ void wmma_store(const v8f& c, FS fs) {
  const int lane = threadIdx.x & 31;
  const int half = lane >> 4;
  const int l16  = lane & 15;
#pragma unroll
  for (int v = 0; v < 8; ++v) fs(v + 8 * half, l16, c[v]);
}

__device__ __forceinline__ float gelu_(float x) {
  return 0.5f * x * (1.f + erff(x * 0.70710678118654752f));
}

// ---------------------------------------------------------------------------
// Unified conv 3x3 implicit GEMM (used for all group convs and the depth
// conv slices). The B operand (input patch) is staged zero-padded into LDS
// per 32-channel K-chunk, so the hot fragment fetch is an unconditional
// ds_load. grid(15, B), block 128 (4 waves -> 64 output channels).
//   out[b*out_bs + (out_coff + m*out_cs)*225 + n]
// ---------------------------------------------------------------------------
__global__ void __launch_bounds__(128)
conv2d_kernel(const float* __restrict__ in, int in_bs, int cin_off, int Cin,
              const float* __restrict__ w,   // (64, Cin, 3, 3) row-major
              const float* __restrict__ bias,
              const float* __restrict__ bng, const float* __restrict__ bnb,
              int use_bn,
              float* __restrict__ out, int out_bs, int out_coff, int out_cs) {
  __shared__ float pad[32 * 68];   // 32 ch x 4 rows x 17 cols, zero-padded
  const int b = blockIdx.y;
  const int ntile = blockIdx.x;
  const int mtile = threadIdx.x >> 5;
  const int lane = threadIdx.x & 31;
  const int tid = threadIdx.x;
  const int K9 = Cin * 9;
  // per-lane output pixel (B-fragment column == this lane's l16)
  const int nn = ntile * 16 + (lane & 15);
  const int sp = nn < 225 ? nn : 224;
  const int oh = sp / 15, ow = sp - oh * 15;
  const int oh0 = (ntile * 16) / 15;       // first output row of this tile
  const int rbase = oh - (oh0 - 1);        // 1 or 2
  const float* inb = in + (size_t)b * in_bs + (size_t)cin_off * 225;
  __builtin_prefetch(w + (size_t)(mtile * 16) * K9, 0, 3);
  v8f acc = {};
  for (int k0 = 0; k0 < Cin; k0 += 32) {
    __syncthreads();
    // cooperative zero-padded stage of 32 channels x (4 x 17) patch
    for (int i = tid; i < 32 * 68; i += 128) {
      const int ci = i / 68, rem = i - ci * 68;
      const int rr = rem / 17, cc = rem - rr * 17;
      const int ih = oh0 - 1 + rr, iw = cc - 1;
      float v = 0.f;
      if ((k0 + ci) < Cin && (unsigned)ih < 15u && (unsigned)iw < 15u)
        v = inb[(size_t)(k0 + ci) * 225 + ih * 15 + iw];
      pad[i] = v;
    }
    __syncthreads();
#pragma unroll
    for (int pos = 0; pos < 9; ++pos) {
      const int kh = pos / 3, kw = pos - kh * 3;
      const int off = (rbase + kh - 1) * 17 + (ow + kw);   // in [0, 68)
      acc = wmma_mac(acc, 32,
        [&](int r, int k) -> float {
          const int kg = k0 + k;
          const int kk = kg < Cin ? kg : 0;
          const float v = w[(size_t)(mtile * 16 + r) * K9 + kk * 9 + pos];
          return kg < Cin ? v : 0.f;
        },
        [&](int k, int) -> float {
          return pad[k * 68 + off];   // unconditional, zero-padded LDS
        });
    }
  }
  const float sc = rsqrtf(1.f + EPSF);
  wmma_store(acc, [&](int mr, int nc, float val) {
    const int m = mtile * 16 + mr;
    const int n2 = ntile * 16 + nc;
    if (n2 < 225) {
      float o = val + bias[m];
      if (use_bn) o = o * (bng[m] * sc) + bnb[m];
      out[(size_t)b * out_bs + (size_t)(out_coff + m * out_cs) * 225 + n2] = o;
    }
  });
}

// ---------------------------------------------------------------------------
// ln_s: normalize per (b, f) over (NS=9, 25, 32) = 7200 elems, write to
// S[b][ns][f][p][pf] layout. grid(10, B), block 256.
// ---------------------------------------------------------------------------
__global__ void __launch_bounds__(256)
ln_s_kernel(const float* __restrict__ dconv, const float* __restrict__ g,
            const float* __restrict__ bb, float* __restrict__ Sout) {
  const int f = blockIdx.x, b = blockIdx.y, tid = threadIdx.x;
  __shared__ float rs[256], rq[256];
  __shared__ float smv, siv;
  float s = 0.f, q = 0.f;
  for (int i = tid; i < 7200; i += 256) {
    const int sp = i >> 5, pf = i & 31;
    const float v = dconv[(size_t)b * 72000 + (f * 32 + pf) * 225 + sp];
    s += v; q += v * v;
  }
  rs[tid] = s; rq[tid] = q; __syncthreads();
  for (int off = 128; off; off >>= 1) {
    if (tid < off) { rs[tid] += rs[tid + off]; rq[tid] += rq[tid + off]; }
    __syncthreads();
  }
  if (!tid) {
    const float m = rs[0] / 7200.f;
    smv = m;
    siv = rsqrtf(fmaxf(rq[0] / 7200.f - m * m, 0.f) + EPSF);
  }
  __syncthreads();
  const float m = smv, inv = siv;
  for (int i = tid; i < 7200; i += 256) {
    const int sp = i >> 5, pf = i & 31;
    const float v = dconv[(size_t)b * 72000 + (f * 32 + pf) * 225 + sp];
    const int hh = sp / 15, ww = sp - hh * 15;
    const int nh = hh / 5, ph = hh - nh * 5, nw = ww / 5, pw = ww - nw * 5;
    const int ns_ = nh * 3 + nw, p = ph * 5 + pw;
    const int gi = ns_ * 800 + p * 32 + pf;
    Sout[(size_t)(((b * 9 + ns_) * 10) + f) * 800 + p * 32 + pf] =
        (v - m) * inv * g[gi] + bb[gi];
  }
}

// ---------------------------------------------------------------------------
// Recurrent transformer cell (shared by short and long stages).
// ---------------------------------------------------------------------------
struct Cell {
  const float *alpha, *cur_b, *cur_w, *ln1_b, *ln1_g, *ln2_b, *ln2_g,
              *pre_b, *pre_w, *proj_b, *proj_w,
              *l1_b, *l1_w, *l2_b, *l2_w, *fln_b, *fln_g,
              *lnc_b, *lnc_g, *lno_b, *lno_g, *lnp_b, *lnp_g;
};

struct CellShm {
  float h[32 * 32];    // recurrent state
  float x[32 * 32];    // ln_cur(x)
  float hl[32 * 32];   // ln_pre(h), later o_hat
  float qc[32 * 96];   // qkv(cur), then blended qkv
  float qp[32 * 96];   // qkv(pre), then attn*v / proj scratch
  float w[32 * 32];    // attention scores
  float mid[32 * 112]; // ffn hidden (padded to 7 N-tiles)
};

__device__ void cell_step(const Cell& C, CellShm& S, int T, int D, int FFD,
                          const float* __restrict__ xin, int x_ts, int x_js,
                          float* __restrict__ outp, int o_ts) {
  const int tid = threadIdx.x;
  const int NT = blockDim.x;
  const int wave = tid >> 5;
  const int nw = NT >> 5;
  const int threeD = 3 * D;

  // ln_cur(x) -> S.x ; ln_pre(h) -> S.hl
  for (int t = tid; t < T; t += NT) {
    float s = 0.f, q = 0.f;
    for (int j = 0; j < D; ++j) { float v = xin[t * x_ts + j * x_js]; s += v; q += v * v; }
    float m = s / D, inv = rsqrtf(fmaxf(q / D - m * m, 0.f) + EPSF);
    for (int j = 0; j < D; ++j)
      S.x[t * 32 + j] = (xin[t * x_ts + j * x_js] - m) * inv * C.lnc_g[j] + C.lnc_b[j];
    s = q = 0.f;
    for (int j = 0; j < D; ++j) { float v = S.h[t * 32 + j]; s += v; q += v * v; }
    m = s / D; inv = rsqrtf(fmaxf(q / D - m * m, 0.f) + EPSF);
    for (int j = 0; j < D; ++j)
      S.hl[t * 32 + j] = (S.h[t * 32 + j] - m) * inv * C.lnp_g[j] + C.lnp_b[j];
  }
  __syncthreads();

  // qkv projections for cur and pre (stores unconditional into padded LDS)
  const int nt3 = (threeD + 15) / 16;
  for (int tile = wave; tile < 2 * nt3; tile += nw) {
    const int tm = tile & 1, tn = tile >> 1;
    v8f c = wmma_acc(D,
      [&](int r, int k) -> float {
        const float v = S.x[(tm * 16 + r) * 32 + k];
        return ((tm * 16 + r) < T && k < D) ? v : 0.f;
      },
      [&](int k, int n) -> float {
        const int col = tn * 16 + n;
        const int cc = col < threeD ? col : 0;
        const int kk = k < D ? k : 0;
        const float v = C.cur_w[cc * D + kk];
        return (col < threeD && k < D) ? v : 0.f;
      });
    wmma_store(c, [&](int mr, int nc, float v) {
      const int col = tn * 16 + nc;
      const int cc = col < threeD ? col : 0;
      S.qc[(tm * 16 + mr) * 96 + col] = v + C.cur_b[cc];
    });
    v8f c2 = wmma_acc(D,
      [&](int r, int k) -> float {
        const float v = S.hl[(tm * 16 + r) * 32 + k];
        return ((tm * 16 + r) < T && k < D) ? v : 0.f;
      },
      [&](int k, int n) -> float {
        const int col = tn * 16 + n;
        const int cc = col < threeD ? col : 0;
        const int kk = k < D ? k : 0;
        const float v = C.pre_w[cc * D + kk];
        return (col < threeD && k < D) ? v : 0.f;
      });
    wmma_store(c2, [&](int mr, int nc, float v) {
      const int col = tn * 16 + nc;
      const int cc = col < threeD ? col : 0;
      S.qp[(tm * 16 + mr) * 96 + col] = v + C.pre_b[cc];
    });
  }
  __syncthreads();

  // sigmoid(alpha) blend into S.qc
  for (int i = tid; i < T * 96; i += NT) {
    const int t = i / 96, j = i - t * 96;
    if (j < threeD) {
      const int third = (j >= 2 * D) ? 2 : (j >= D) ? 1 : 0;
      const int jd = j - third * D;
      const float al = 1.f / (1.f + expf(-C.alpha[third * D + jd]));
      S.qc[t * 96 + j] = al * S.qc[t * 96 + j] + (1.f - al) * S.qp[t * 96 + j];
    }
  }
  __syncthreads();

  // scores = q k^T * D^-0.5
  const float scl = rsqrtf((float)D);
  const int ntT = (T + 15) / 16;
  for (int tile = wave; tile < 2 * ntT; tile += nw) {
    const int tm = tile & 1, tn = tile >> 1;
    v8f c = wmma_acc(D,
      [&](int r, int k) -> float {
        const float v = S.qc[(tm * 16 + r) * 96 + k];
        return ((tm * 16 + r) < T && k < D) ? v : 0.f;
      },
      [&](int k, int n) -> float {
        const int col = tn * 16 + n;
        const float v = S.qc[(col & 31) * 96 + D + k];
        return (col < T && k < D) ? v : 0.f;
      });
    wmma_store(c, [&](int mr, int nc, float v) {
      S.w[(tm * 16 + mr) * 32 + tn * 16 + nc] = v * scl;
    });
  }
  __syncthreads();

  // softmax rows
  for (int t = tid; t < T; t += NT) {
    float mx = -1e30f;
    for (int n = 0; n < T; ++n) mx = fmaxf(mx, S.w[t * 32 + n]);
    float sm = 0.f;
    for (int n = 0; n < T; ++n) { float e = expf(S.w[t * 32 + n] - mx); S.w[t * 32 + n] = e; sm += e; }
    const float r = 1.f / sm;
    for (int n = 0; n < T; ++n) S.w[t * 32 + n] *= r;
  }
  __syncthreads();

  // av = softmax @ v -> S.qp[0:1024)
  const int ntD = (D + 15) / 16;
  for (int tile = wave; tile < 2 * ntD; tile += nw) {
    const int tm = tile & 1, tn = tile >> 1;
    v8f c = wmma_acc(T,
      [&](int r, int k) -> float {
        const float v = S.w[(tm * 16 + r) * 32 + (k & 31)];
        return ((tm * 16 + r) < T && k < T) ? v : 0.f;
      },
      [&](int k, int n) -> float {
        const int col = tn * 16 + n;
        const float v = S.qc[(k & 31) * 96 + 2 * D + (col & 31)];
        return (k < T && col < D) ? v : 0.f;
      });
    wmma_store(c, [&](int mr, int nc, float v) {
      S.qp[(tm * 16 + mr) * 32 + tn * 16 + nc] = v;
    });
  }
  __syncthreads();

  // ln1 in place
  for (int t = tid; t < T; t += NT) {
    float s = 0.f, q = 0.f;
    for (int j = 0; j < D; ++j) { float v = S.qp[t * 32 + j]; s += v; q += v * v; }
    const float m = s / D, inv = rsqrtf(fmaxf(q / D - m * m, 0.f) + EPSF);
    for (int j = 0; j < D; ++j)
      S.qp[t * 32 + j] = (S.qp[t * 32 + j] - m) * inv * C.ln1_g[j] + C.ln1_b[j];
  }
  __syncthreads();

  // proj -> po
  float* po = S.qp + 1024;
  for (int tile = wave; tile < 2 * ntD; tile += nw) {
    const int tm = tile & 1, tn = tile >> 1;
    v8f c = wmma_acc(D,
      [&](int r, int k) -> float {
        const float v = S.qp[(tm * 16 + r) * 32 + k];
        return ((tm * 16 + r) < T && k < D) ? v : 0.f;
      },
      [&](int k, int n) -> float {
        const int col = tn * 16 + n;
        const int cc = col < D ? col : 0;
        const int kk = k < D ? k : 0;
        const float v = C.proj_w[cc * D + kk];
        return (col < D && k < D) ? v : 0.f;
      });
    wmma_store(c, [&](int mr, int nc, float v) {
      const int col = tn * 16 + nc;
      const int cc = col < D ? col : 0;
      po[(tm * 16 + mr) * 32 + col] = v + C.proj_b[cc];
    });
  }
  __syncthreads();

  // ln2, a = o2 + relu(x_ln); h_new = h_ln + a; o_hat = ln_o(x_ln + a)
  for (int t = tid; t < T; t += NT) {
    float s = 0.f, q = 0.f;
    for (int j = 0; j < D; ++j) { float v = po[t * 32 + j]; s += v; q += v * v; }
    float m = s / D, inv = rsqrtf(fmaxf(q / D - m * m, 0.f) + EPSF);
    float s2 = 0.f, q2 = 0.f;
    for (int j = 0; j < D; ++j) {
      const float o2 = (po[t * 32 + j] - m) * inv * C.ln2_g[j] + C.ln2_b[j];
      const float xv = S.x[t * 32 + j];
      const float a_ = o2 + fmaxf(xv, 0.f);
      S.h[t * 32 + j] = S.hl[t * 32 + j] + a_;
      const float xa = xv + a_;
      po[t * 32 + j] = xa;
      s2 += xa; q2 += xa * xa;
    }
    m = s2 / D; inv = rsqrtf(fmaxf(q2 / D - m * m, 0.f) + EPSF);
    for (int j = 0; j < D; ++j)
      S.hl[t * 32 + j] = (po[t * 32 + j] - m) * inv * C.lno_g[j] + C.lno_b[j];
  }
  __syncthreads();
  if (!outp) return;   // recurrence only needs h

  // ffn l1
  const int ntF = (FFD + 15) / 16;
  for (int tile = wave; tile < 2 * ntF; tile += nw) {
    const int tm = tile & 1, tn = tile >> 1;
    v8f c = wmma_acc(D,
      [&](int r, int k) -> float {
        const float v = S.hl[(tm * 16 + r) * 32 + k];
        return ((tm * 16 + r) < T && k < D) ? v : 0.f;
      },
      [&](int k, int n) -> float {
        const int col = tn * 16 + n;
        const int cc = col < FFD ? col : 0;
        const int kk = k < D ? k : 0;
        const float v = C.l1_w[cc * D + kk];
        return (col < FFD && k < D) ? v : 0.f;
      });
    wmma_store(c, [&](int mr, int nc, float v) {
      const int col = tn * 16 + nc;
      const int cc = col < FFD ? col : 0;
      S.mid[(tm * 16 + mr) * 112 + col] = v + C.l1_b[cc];
    });
  }
  __syncthreads();

  // ffn ln + gelu
  for (int t = tid; t < T; t += NT) {
    float s = 0.f, q = 0.f;
    for (int j = 0; j < FFD; ++j) { float v = S.mid[t * 112 + j]; s += v; q += v * v; }
    const float m = s / FFD, inv = rsqrtf(fmaxf(q / FFD - m * m, 0.f) + EPSF);
    for (int j = 0; j < FFD; ++j) {
      const float v = (S.mid[t * 112 + j] - m) * inv * C.fln_g[j] + C.fln_b[j];
      S.mid[t * 112 + j] = gelu_(v);
    }
  }
  __syncthreads();

  // out = mid @ l2^T + b2 + elu(o_hat)
  for (int tile = wave; tile < 2 * ntD; tile += nw) {
    const int tm = tile & 1, tn = tile >> 1;
    v8f c = wmma_acc(FFD,
      [&](int r, int k) -> float {
        const int kk = k < FFD ? k : 0;
        const float v = S.mid[(tm * 16 + r) * 112 + kk];
        return ((tm * 16 + r) < T && k < FFD) ? v : 0.f;
      },
      [&](int k, int n) -> float {
        const int col = tn * 16 + n;
        const int cc = col < D ? col : 0;
        const int kk = k < FFD ? k : 0;
        const float v = C.l2_w[cc * FFD + kk];
        return (col < D && k < FFD) ? v : 0.f;
      });
    wmma_store(c, [&](int mr, int nc, float v) {
      const int row = tm * 16 + mr, col = tn * 16 + nc;
      if (row < T && col < D) {
        const float oh = S.hl[row * 32 + col];
        const float e = oh > 0.f ? oh : (expf(oh) - 1.f);
        outp[row * o_ts + col] = v + C.l2_b[col] + e;
      }
    });
  }
  __syncthreads();
}

// Short stage: grid(9, B), loops the 10-step recurrence in-kernel.
__global__ void __launch_bounds__(256)
short_kernel(Cell c0, const float* __restrict__ Sin, float* __restrict__ sout) {
  __shared__ CellShm shm;
  const int ns = blockIdx.x, b = blockIdx.y;
  Cell C = c0;   // apply stacked (NS-leading) offsets, d=32 ffd=64
  C.alpha += ns * 96;  C.cur_b += ns * 96;  C.cur_w += ns * 3072;
  C.ln1_b += ns * 32;  C.ln1_g += ns * 32;  C.ln2_b += ns * 32;  C.ln2_g += ns * 32;
  C.pre_b += ns * 96;  C.pre_w += ns * 3072; C.proj_b += ns * 32; C.proj_w += ns * 1024;
  C.l1_b  += ns * 64;  C.l1_w  += ns * 2048; C.l2_b  += ns * 32;  C.l2_w  += ns * 2048;
  C.fln_b += ns * 64;  C.fln_g += ns * 64;
  C.lnc_b += ns * 32;  C.lnc_g += ns * 32;  C.lno_b += ns * 32;  C.lno_g += ns * 32;
  C.lnp_b += ns * 32;  C.lnp_g += ns * 32;
  __builtin_prefetch(C.cur_w, 0, 3);
  __builtin_prefetch(C.pre_w, 0, 3);
  for (int i = threadIdx.x; i < 1024; i += 256) shm.h[i] = 0.f;
  __syncthreads();
  for (int f = 0; f < 10; ++f) {
    const float* xin = Sin + (size_t)((b * 9 + ns) * 10 + f) * 800;
    cell_step(C, shm, 25, 32, 64, xin, 32, 1,
              (f == 9) ? sout + (size_t)(b * 9 + ns) * 800 : nullptr, 32);
  }
}

// Long stage: grid(2, B) (dir 0=forward, 1=reverse), 9 steps in-kernel.
__global__ void __launch_bounds__(256)
long_kernel(Cell cf, Cell cr, const float* __restrict__ sbuf,
            float* __restrict__ xbuf) {
  __shared__ CellShm shm;
  const int dir = blockIdx.x, b = blockIdx.y;
  const Cell C = dir ? cr : cf;
  __builtin_prefetch(C.cur_w, 0, 3);
  __builtin_prefetch(C.pre_w, 0, 3);
  for (int i = threadIdx.x; i < 1024; i += 256) shm.h[i] = 0.f;
  __syncthreads();
  for (int s = 0; s < 9; ++s) {
    const int ns = dir ? (8 - s) : s;
    // x[t][j] = short[b][ns][j][t]  (transposed view)
    const float* xin = sbuf + (size_t)(b * 9 + ns) * 800;
    cell_step(C, shm, 32, 25, 100, xin, 1, 32,
              xbuf + (size_t)((b * 2 + dir) * 9 + s) * 800, 25);
  }
}

// ln_f / ln_r over (9,32,25) per b; emit only ns=8 slice into cls.
__global__ void __launch_bounds__(256)
ln_cls_kernel(const float* __restrict__ xbuf, const float* __restrict__ gf,
              const float* __restrict__ bf, const float* __restrict__ gr,
              const float* __restrict__ br, float* __restrict__ cls) {
  const int dir = blockIdx.x, b = blockIdx.y, tid = threadIdx.x;
  const float* base = xbuf + (size_t)(b * 2 + dir) * 7200;
  __shared__ float rs[256], rq[256];
  __shared__ float smv, siv;
  float s = 0.f, q = 0.f;
  for (int i = tid; i < 7200; i += 256) { float v = base[i]; s += v; q += v * v; }
  rs[tid] = s; rq[tid] = q; __syncthreads();
  for (int off = 128; off; off >>= 1) {
    if (tid < off) { rs[tid] += rs[tid + off]; rq[tid] += rq[tid + off]; }
    __syncthreads();
  }
  if (!tid) {
    const float m = rs[0] / 7200.f;
    smv = m;
    siv = rsqrtf(fmaxf(rq[0] / 7200.f - m * m, 0.f) + EPSF);
  }
  __syncthreads();
  const float* g = dir ? gr : gf;
  const float* bb = dir ? br : bf;
  for (int i = tid; i < 800; i += 256) {
    const float v = base[8 * 800 + i];
    cls[(size_t)b * 1600 + dir * 800 + i] =
        (v - smv) * siv * g[8 * 800 + i] + bb[8 * 800 + i];
  }
}

// head: (bn -> l1) as WMMA GEMM, M=256 rows, K=1600, N=100. grid(7,16), block 32.
__global__ void head_l1_kernel(const float* __restrict__ cls,
                               const float* __restrict__ bng,
                               const float* __restrict__ bnb,
                               const float* __restrict__ w1,
                               const float* __restrict__ b1,
                               float* __restrict__ h1) {
  const int mtile = blockIdx.y, ntile = blockIdx.x;
  const float sc = rsqrtf(1.f + EPSF);
  __builtin_prefetch(w1 + (size_t)(ntile * 16) * 1600, 0, 3);
  v8f c = wmma_acc(1600,
    [&](int r, int k) -> float {
      const int row = mtile * 16 + r;
      return cls[(size_t)row * 1600 + k] * (bng[k] * sc) + bnb[k];
    },
    [&](int k, int n) -> float {
      const int col = ntile * 16 + n;
      const int cc = col < 100 ? col : 0;
      const float v = w1[(size_t)cc * 1600 + k];
      return col < 100 ? v : 0.f;
    });
  wmma_store(c, [&](int mr, int nc, float v) {
    const int row = mtile * 16 + mr, col = ntile * 16 + nc;
    if (col < 100) h1[row * 100 + col] = v + b1[col];
  });
}

// head tail: l2 (100->64) + ln + gelu + l3 (64->16). grid(B), block 64.
__global__ void __launch_bounds__(64)
head_tail_kernel(const float* __restrict__ h1, const float* __restrict__ w2,
                 const float* __restrict__ b2, const float* __restrict__ lng,
                 const float* __restrict__ lnb, const float* __restrict__ w3,
                 const float* __restrict__ b3, float* __restrict__ outp) {
  const int b = blockIdx.x, j = threadIdx.x;
  __shared__ float sy[64], rs[64], rq[64];
  __shared__ float smv, siv;
  float acc = b2[j];
  for (int k = 0; k < 100; ++k) acc += h1[b * 100 + k] * w2[j * 100 + k];
  rs[j] = acc; rq[j] = acc * acc; __syncthreads();
  for (int off = 32; off; off >>= 1) {
    if (j < off) { rs[j] += rs[j + off]; rq[j] += rq[j + off]; }
    __syncthreads();
  }
  if (!j) {
    const float m = rs[0] / 64.f;
    smv = m;
    siv = rsqrtf(fmaxf(rq[0] / 64.f - m * m, 0.f) + EPSF);
  }
  __syncthreads();
  const float v = (acc - smv) * siv * lng[j] + lnb[j];
  sy[j] = gelu_(v);
  __syncthreads();
  if (j < 16) {
    float o = b3[j];
    for (int t = 0; t < 64; ++t) o += sy[t] * w3[j * 64 + t];
    outp[b * 16 + j] = o;
  }
}

// ---------------------------------------------------------------------------
// Host side: map the 127 param leaves (jax pytree order: sorted dict keys)
// plus x (position auto-detected by its unique size), then launch the pipe.
// ---------------------------------------------------------------------------
static Cell make_cell(const float* const* L, int base) {
  Cell c;
  c.alpha = L[base + 0];  c.cur_b = L[base + 1];  c.cur_w = L[base + 2];
  c.ln1_b = L[base + 3];  c.ln1_g = L[base + 4];  c.ln2_b = L[base + 5];
  c.ln2_g = L[base + 6];  c.pre_b = L[base + 7];  c.pre_w = L[base + 8];
  c.proj_b = L[base + 9]; c.proj_w = L[base + 10];
  c.l1_b = L[base + 11];  c.l1_w = L[base + 12];  c.l2_b = L[base + 13];
  c.l2_w = L[base + 14];  c.fln_b = L[base + 15]; c.fln_g = L[base + 16];
  c.lnc_b = L[base + 17]; c.lnc_g = L[base + 18]; c.lno_b = L[base + 19];
  c.lno_g = L[base + 20]; c.lnp_b = L[base + 21]; c.lnp_g = L[base + 22];
  return c;
}

extern "C" void kernel_launch(void* const* d_in, const int* in_sizes, int n_in,
                              void* d_out, int out_size, void* d_ws,
                              size_t ws_size, hipStream_t stream) {
  (void)out_size; (void)ws_size;
  const float* X;
  const float* L[127];
  const long long XSZ = 256LL * 200 * 15 * 15; // 11,520,000 - unique size
  if (n_in >= 1 && (long long)in_sizes[0] == XSZ) {
    X = (const float*)d_in[0];
    for (int i = 0; i < 127 && i + 1 < n_in; ++i) L[i] = (const float*)d_in[i + 1];
  } else {
    for (int i = 0; i < 127 && i < n_in; ++i) L[i] = (const float*)d_in[i];
    X = (const float*)d_in[n_in - 1];
  }

  float* ws = (float*)d_ws;
  float* grp    = ws;                               // B*64*225
  float* cat    = grp + (size_t)256 * 64 * 225;     // B*320*225
  float* dcv    = cat + (size_t)256 * 320 * 225;    // B*320*225
  float* Sbuf   = cat;                              // reuse cat after depth conv
  float* sshort = dcv + (size_t)256 * 320 * 225;    // B*9*800
  float* xlr    = sshort + (size_t)256 * 9 * 800;   // B*2*9*800
  float* cls    = xlr + (size_t)256 * 2 * 9 * 800;  // B*1600
  float* h1     = cls + (size_t)256 * 1600;         // B*100

  static const int GS[5] = {0, 40, 80, 105, 145};
  static const int GC[5] = {40, 40, 25, 40, 55};
  for (int g = 0; g < 5; ++g) {
    // group leaf order (sorted keys): b1,b2,bn1.b,bn1.g,bn2.b,bn2.g,w1,w2
    const float* b1   = L[g * 8 + 0];
    const float* b2   = L[g * 8 + 1];
    const float* bn1b = L[g * 8 + 2];
    const float* bn1g = L[g * 8 + 3];
    const float* bn2b = L[g * 8 + 4];
    const float* bn2g = L[g * 8 + 5];
    const float* w1   = L[g * 8 + 6];
    const float* w2   = L[g * 8 + 7];
    conv2d_kernel<<<dim3(15, 256), 128, 0, stream>>>(
        X, 200 * 225, GS[g], GC[g], w1, b1, bn1g, bn1b, 1,
        grp, 64 * 225, 0, 1);
    conv2d_kernel<<<dim3(15, 256), 128, 0, stream>>>(
        grp, 64 * 225, 0, 64, w2, b2, bn2g, bn2b, 1,
        cat, 320 * 225, g * 64, 1);
  }
  // depth conv: one launch per depth slice dz; channel c = o*5+dz.
  // depth.b = L[40], depth.w = L[41]
  for (int dz = 0; dz < 5; ++dz) {
    conv2d_kernel<<<dim3(15, 256), 128, 0, stream>>>(
        cat, 320 * 225, dz * 64, 64, L[41], L[40], nullptr, nullptr, 0,
        dcv, 320 * 225, dz, 5);
  }
  // ln_s: b=L[56], g=L[57]
  ln_s_kernel<<<dim3(10, 256), 256, 0, stream>>>(dcv, L[57], L[56], Sbuf);

  Cell cshort = make_cell(L, 104);   // 'short' (stacked NS)
  short_kernel<<<dim3(9, 256), 256, 0, stream>>>(cshort, Sbuf, sshort);

  Cell cf = make_cell(L, 58);        // 'long_f'
  Cell cr = make_cell(L, 81);        // 'long_r'
  long_kernel<<<dim3(2, 256), 256, 0, stream>>>(cf, cr, sshort, xlr);

  // ln_f: b=L[52], g=L[53]; ln_r: b=L[54], g=L[55]
  ln_cls_kernel<<<dim3(2, 256), 256, 0, stream>>>(xlr, L[53], L[52], L[55],
                                                  L[54], cls);
  // head: bn.b=42 bn.g=43 l1.b=44 l1.w=45 l2.b=46 l2.w=47 l3.b=48 l3.w=49
  //       ln.b=50 ln.g=51
  head_l1_kernel<<<dim3(7, 16), 32, 0, stream>>>(cls, L[43], L[42], L[45],
                                                 L[44], h1);
  head_tail_kernel<<<dim3(256), 64, 0, stream>>>(h1, L[47], L[46], L[51],
                                                 L[50], L[49], L[48],
                                                 (float*)d_out);
}
